// FullRankRNN_74337293959603
// MI455X (gfx1250) — matrix operand order repaired and checked by reading, not compiled, not run on hardware
//
#include <hip/hip_runtime.h>

// FullRankRNN forward, MI455X (gfx1250, wave32, WMMA).
// Pass 1 (parallel GEMM): pre[B,T,H] = ALPHA*(input @ (wi*si)) + NOISE_STD*noise  -> d_ws
// Pass 2 (persistent):    h_{t+1} = (1-ALPHA)h + ALPHA*(relu(h) @ wrec.T) + pre[t]
//                         out_t   = relu(h_{t+1}) @ (wo*so)
// One workgroup per 16-row batch tile, 16 waves, one barrier per step.
// wrec f16 split: K-tiles 0..7 as LDS B-fragments (256KB), 8..15 register-resident.

#define ALPHA_    0.2f
#define NOISESTD_ 0.05f

constexpr int B_ = 128, T_ = 1024, I_ = 64, H_ = 512, O_ = 10;
constexpr int BT      = 16;            // batch tile (WMMA M)
constexpr int NWAVES  = 16;            // waves per workgroup
constexpr int THREADS = NWAVES * 32;   // 512

// Main-kernel LDS layout (units: _Float16 elements)
//  [0, 131072)        wrecLo B-frags: frag f = ntile*8 + kt (kt 0..7), 512 halves each
//  [131072, 139264)   r2 buffer 0 : 16 A-frag tiles * 512 halves
//  [139264, 147456)   r2 buffer 1
//  [147456, 155648)   wo B-frags : 16 tiles * 512 halves
//  byte 311296..313344  out accumulators: 2 * 16*16 f32
constexpr int WRECLO_OFF = 0;
constexpr int R2_OFF     = 131072;
constexpr int WO_OFF     = 147456;
constexpr int OUT_BYTE   = 311296;
constexpr int SMEM_BYTES = 313344;

typedef __attribute__((ext_vector_type(16))) _Float16 v16h;
typedef __attribute__((ext_vector_type(8)))  _Float16 v8h;
typedef __attribute__((ext_vector_type(8)))  float    v8f;

// A-matrix fragment slot for element (m, kk) within a 16x32 f16 tile.
__device__ __forceinline__ int a_pos(int m, int kk) {
  int lr = m + (((kk >> 4) & 1) << 4);
  int s  = kk & 15;
  return lr * 16 + s;
}

// Load one 16-half fragment (A or B) for this lane: two ds_load_b128.
__device__ __forceinline__ v16h load_frag(const _Float16* tileBase, int lane) {
  const v8h* p = (const v8h*)(tileBase + lane * 16);
  v8h lo = p[0];
  v8h hi = p[1];
  return __builtin_shufflevector(lo, hi, 0,1,2,3,4,5,6,7,8,9,10,11,12,13,14,15);
}

__device__ __forceinline__ v8f wmma_f16(v16h a, v16h b, v8f c) {
  return __builtin_amdgcn_wmma_f32_16x16x32_f16(false, a, false, b, (short)0, c,
                                                false, false);
}

// ============================================================================
// Pass 1: pre = ALPHA * (input @ wi_full) + NOISE_STD * noise
// Grid: B*T/16 blocks; block = 16 waves; wave w does column tiles w and w+16.
// ============================================================================
__global__ __launch_bounds__(THREADS)
void xproj_kernel(const float* __restrict__ input,  // [B*T, I]
                  const float* __restrict__ noise,  // [B*T, H]
                  const float* __restrict__ wi,     // [I, H]
                  const float* __restrict__ si,     // [I]
                  float* __restrict__ pre)          // [B*T, H]
{
  __shared__ _Float16 xa[1024];        // 2 A-frag tiles (16 x 64)

  const int tid  = (int)threadIdx.x;
  const int lane = tid & 31;
  const int wv   = tid >> 5;
  const int nloc = lane & 15;
  const int mhi  = (lane >> 4) << 3;
  const int khlf = (lane >> 4) << 4;
  const int row0 = (int)blockIdx.x * BT;   // row in flattened [B*T]

  // stage x rows as A-fragments (f16)
  #pragma unroll
  for (int r = 0; r < 2; ++r) {
    int e = tid + r * THREADS;           // 16x64
    int m = e >> 6, i = e & 63;
    xa[(i >> 5) * 512 + a_pos(m, i & 31)] = (_Float16)input[(size_t)(row0 + m) * I_ + i];
  }

  // wi_full B-fragments for this wave's two column tiles
  v16h wiF[2][2];
  #pragma unroll
  for (int ct = 0; ct < 2; ++ct) {
    const int j = (wv + ct * 16) * 16 + nloc;
    #pragma unroll
    for (int kt = 0; kt < 2; ++kt) {
      v16h v;
      #pragma unroll
      for (int s = 0; s < 16; ++s) {
        int i = kt * 32 + khlf + s;
        v[s] = (_Float16)(wi[i * H_ + j] * si[i]);
      }
      wiF[kt][ct] = v;
    }
  }

  __syncthreads();

  v16h ax0 = load_frag(xa, lane);
  v16h ax1 = load_frag(xa + 512, lane);
  #pragma unroll
  for (int ct = 0; ct < 2; ++ct) {
    v8f z = {};
    v8f acc = wmma_f16(ax0, wiF[0][ct], z);
    acc     = wmma_f16(ax1, wiF[1][ct], acc);
    const int col = (wv + ct * 16) * 16 + nloc;
    #pragma unroll
    for (int e = 0; e < 8; ++e) {
      size_t idx = (size_t)(row0 + e + mhi) * H_ + col;
      pre[idx] = ALPHA_ * acc[e] + NOISESTD_ * noise[idx];
    }
  }
}

// ============================================================================
// Pass 2: persistent recurrence
// ============================================================================
__global__ __launch_bounds__(THREADS)
void rnn_persistent_kernel(const float* __restrict__ pre,   // [B,T,H]
                           const float* __restrict__ wrec,  // [H,H]
                           const float* __restrict__ wo,    // [H,O]
                           const float* __restrict__ so,    // [O]
                           const float* __restrict__ h0,    // [H]
                           float* __restrict__ out)         // [B,T,O]
{
  extern __shared__ _Float16 smem[];
  _Float16* wrecLo = smem + WRECLO_OFF;
  _Float16* r2b0   = smem + R2_OFF;
  _Float16* r2b1   = smem + R2_OFF + 8192;
  _Float16* woLds  = smem + WO_OFF;
  float*    outAcc = (float*)((char*)smem + OUT_BYTE); // [2][16][16]

  const int tid   = (int)threadIdx.x;
  const int lane  = tid & 31;
  const int wv    = tid >> 5;           // 0..15
  const int nloc  = lane & 15;
  const int mhi   = (lane >> 4) << 3;   // +8 row offset (C/D layout, hi half-wave)
  const int bbase = (int)blockIdx.x * BT;

  // ---- zero the output accumulators ----
  for (int i = tid; i < 512; i += THREADS) outAcc[i] = 0.0f;

  // ---- stage wrec K-tiles 0..7 into LDS as B-fragments ----
  for (int idx = tid; idx < 8 * 32 * 512; idx += THREADS) {
    int s  = idx & 15;
    int lr = (idx >> 4) & 31;
    int f  = idx >> 9;
    int kt = f & 7;
    int nt = f >> 3;
    int k  = kt * 32 + ((lr >> 4) << 4) + s;   // K index (0..255)
    int j  = nt * 16 + (lr & 15);              // output column
    wrecLo[idx] = (_Float16)wrec[j * H_ + k];  // wrec.T(k,j) = wrec[j,k]
  }

  // ---- stage wo_full B-fragments into LDS (16 K-tiles) ----
  for (int idx = tid; idx < 16 * 512; idx += THREADS) {
    int s  = idx & 15;
    int lr = (idx >> 4) & 31;
    int kt = idx >> 9;
    int k  = kt * 32 + ((lr >> 4) << 4) + s;
    int n  = lr & 15;
    woLds[idx] = (n < O_) ? (_Float16)(wo[k * O_ + n] * so[n]) : (_Float16)0.0f;
  }

  // ---- register-resident wrec B-fragments: K-tiles 8..15 ----
  v16h wrecHi[8][2];
  {
    const int khlf = (lane >> 4) << 4;
    #pragma unroll
    for (int c = 0; c < 2; ++c) {
      const int j = wv * 32 + c * 16 + nloc;
      #pragma unroll
      for (int kt2 = 0; kt2 < 8; ++kt2) {
        const int kbase = (8 + kt2) * 32 + khlf;
        v16h v;
        #pragma unroll
        for (int s = 0; s < 16; ++s) v[s] = (_Float16)wrec[j * H_ + kbase + s];
        wrecHi[kt2][c] = v;
      }
    }
  }

  // ---- hidden state (this wave's 16x32 block) and initial relu(h0) ----
  v8f hreg[2];
  #pragma unroll
  for (int c = 0; c < 2; ++c) {
    const float hv0 = h0[wv * 32 + c * 16 + nloc];
    v8f hv;
    #pragma unroll
    for (int e = 0; e < 8; ++e) hv[e] = hv0;
    hreg[c] = hv;
    const int kk = c * 16 + nloc;
    #pragma unroll
    for (int e = 0; e < 8; ++e) {
      float r = hv0 > 0.0f ? hv0 : 0.0f;
      r2b0[wv * 512 + a_pos(e + mhi, kk)] = (_Float16)r;
    }
  }

  // =======================  main time loop  =======================
  for (int t = 0; t < T_; ++t) {
    const int cur = t & 1;
    _Float16* r2c = cur ? r2b1 : r2b0;
    _Float16* r2n = cur ? r2b0 : r2b1;

    // (1) load pre-projected input+noise for this step (long-latency, pre-barrier)
    float pz[16];
    #pragma unroll
    for (int c = 0; c < 2; ++c) {
      #pragma unroll
      for (int e = 0; e < 8; ++e) {
        int m   = e + mhi;
        int col = wv * 32 + c * 16 + nloc;
        pz[c * 8 + e] = pre[((size_t)(bbase + m) * T_ + t) * H_ + col];
      }
    }

    __syncthreads();

    // (2) drain out_{t-2} from accumulator buffer (t&1), then zero it
    if (t >= 2 && tid < 256) {
      int m = tid >> 4, o = tid & 15;
      float v = outAcc[cur * 256 + tid];
      if (o < O_) out[((size_t)(bbase + m) * T_ + (t - 2)) * O_ + o] = v;
      outAcc[cur * 256 + tid] = 0.0f;
    }

    // (3) acc = relu(h_t) @ wrec.T  -- reg-B and LDS-B chains interleaved
    v8f acc[2][2];
    #pragma unroll
    for (int c = 0; c < 2; ++c) {
      v8f z = {};
      acc[c][0] = z;
      acc[c][1] = z;
    }
    #pragma unroll
    for (int kt = 0; kt < 8; ++kt) {
      v16h a1  = load_frag(r2c + (8 + kt) * 512, lane);   // A for reg-B chain
      v16h a0  = load_frag(r2c + kt * 512, lane);         // A for LDS-B chain
      v16h bL0 = load_frag(wrecLo + ((wv * 2 + 0) * 8 + kt) * 512, lane);
      v16h bL1 = load_frag(wrecLo + ((wv * 2 + 1) * 8 + kt) * 512, lane);
      acc[0][1] = wmma_f16(a1, wrecHi[kt][0], acc[0][1]);
      acc[1][1] = wmma_f16(a1, wrecHi[kt][1], acc[1][1]);
      acc[0][0] = wmma_f16(a0, bL0, acc[0][0]);
      acc[1][0] = wmma_f16(a0, bL1, acc[1][0]);
    }

    // (4) readout partial: out_{t-1} += relu(h_t)[:,Kw] @ wo_full[Kw,:]
    if (t >= 1) {
      v16h a  = load_frag(r2c + wv * 512, lane);
      v16h wf = load_frag(woLds + wv * 512, lane);
      v8f z   = {};
      v8f op  = wmma_f16(a, wf, z);
      float* oa = outAcc + (cur ^ 1) * 256;      // buffer (t-1)&1
      #pragma unroll
      for (int e = 0; e < 8; ++e)
        atomicAdd(&oa[(e + mhi) * 16 + nloc], op[e]);
    }

    // (5) h update + write relu(h_{t+1}) to the other r2 buffer
    #pragma unroll
    for (int c = 0; c < 2; ++c) {
      v8f hv = hreg[c];
      v8f ac = acc[c][0] + acc[c][1];
      const int kk = c * 16 + nloc;
      #pragma unroll
      for (int e = 0; e < 8; ++e) {
        float hn = (1.0f - ALPHA_) * hv[e] + ALPHA_ * ac[e] + pz[c * 8 + e];
        hv[e] = hn;
        float r = hn > 0.0f ? hn : 0.0f;
        r2n[wv * 512 + a_pos(e + mhi, kk)] = (_Float16)r;
      }
      hreg[c] = hv;
    }
  }

  // =======================  epilogue  =======================
  __syncthreads();
  {
    const int cur = T_ & 1;   // 0 for T=1024
    if (tid < 256) {          // drain out_{T-2}
      int m = tid >> 4, o = tid & 15;
      float v = outAcc[cur * 256 + tid];
      if (o < O_) out[((size_t)(bbase + m) * T_ + (T_ - 2)) * O_ + o] = v;
    }
    _Float16* r2c = cur ? r2b1 : r2b0;
    v16h a  = load_frag(r2c + wv * 512, lane);
    v16h wf = load_frag(woLds + wv * 512, lane);
    v8f z   = {};
    v8f op  = wmma_f16(a, wf, z);
    float* oa = outAcc + (cur ^ 1) * 256;
    #pragma unroll
    for (int e = 0; e < 8; ++e)
      atomicAdd(&oa[(e + mhi) * 16 + nloc], op[e]);
  }
  __syncthreads();
  if (tid < 256) {
    int m = tid >> 4, o = tid & 15;
    float v = outAcc[((T_ - 1) & 1) * 256 + tid];
    if (o < O_) out[((size_t)(bbase + m) * T_ + (T_ - 1)) * O_ + o] = v;
  }
}

extern "C" void kernel_launch(void* const* d_in, const int* in_sizes, int n_in,
                              void* d_out, int out_size, void* d_ws, size_t ws_size,
                              hipStream_t stream) {
  const float* input = (const float*)d_in[0];
  const float* noise = (const float*)d_in[1];
  const float* wi    = (const float*)d_in[2];
  const float* si    = (const float*)d_in[3];
  const float* wrec  = (const float*)d_in[4];
  const float* wo    = (const float*)d_in[5];
  const float* so    = (const float*)d_in[6];
  const float* h0    = (const float*)d_in[7];
  float* out = (float*)d_out;
  float* pre = (float*)d_ws;   // needs B*T*H*4 = 268 MB of scratch
  (void)in_sizes; (void)n_in; (void)out_size; (void)ws_size;

  // Pass 1: parallel input projection + noise fold (WMMA GEMM)
  hipLaunchKernelGGL(xproj_kernel, dim3(B_ * T_ / BT), dim3(THREADS), 0, stream,
                     input, noise, wi, si, pre);

  // Pass 2: persistent recurrence (8 workgroups, 16 waves each)
  hipLaunchKernelGGL(rnn_persistent_kernel, dim3(B_ / BT), dim3(THREADS),
                     SMEM_BYTES, stream, pre, wrec, wo, so, h0, out);
}
